// TimeGRUNet_12652973654358
// MI455X (gfx1250) — compile-verified, matrix operand order checked by reading, not run on hardware
//
#include <hip/hip_runtime.h>
#include <hip/hip_bf16.h>
#include <math.h>

// ---------------------------------------------------------------------------
// Types / helpers
// ---------------------------------------------------------------------------
typedef unsigned short u16;
typedef unsigned int   u32;

typedef __attribute__((ext_vector_type(16))) __bf16 v16bf;
typedef __attribute__((ext_vector_type(8)))  float  v8f;
typedef __attribute__((ext_vector_type(4)))  int    v4i;

union FragAB { v16bf v; u32 u[8]; uint4 q[2]; };

__device__ __forceinline__ u16 f2bf(float x) {
  union { float f; u32 u; } c; c.f = x;
  u32 r = c.u + 0x7FFFu + ((c.u >> 16) & 1u);   // round-to-nearest-even
  return (u16)(r >> 16);
}
__device__ __forceinline__ float bf2f(u16 h) {
  union { u32 u; float f; } c; c.u = ((u32)h) << 16;
  return c.f;
}
__device__ __forceinline__ float sigmoidf_(float x) {
  return 1.0f / (1.0f + __expf(-x));
}

// Fragment-major ("swizzled") layout: matrix [R,K] (bf16) is stored as
// row-tiles of 16 x k-blocks of 32 -> 512-element tiles. Within a tile,
// lane l owns elements [l*16, l*16+16) == exactly its WMMA A/B fragment
// (ISA 7.12.2 16-bit operand layout). Fragment load = 2x b128.
__device__ __forceinline__ size_t sw_off(int r, int k, int kb_total) {
  int lane = (r & 15) + (((k >> 3) & 1) << 4);
  int v    = (((k >> 4) & 1) << 2) | ((k >> 1) & 3);
  int e    = (v << 1) | (k & 1);
  return ((size_t)(r >> 4) * kb_total + (k >> 5)) * 512 + lane * 16 + e;
}

// ---------------------------------------------------------------------------
// CDNA5 async global->LDS copy (ASYNCcnt path), with sync fallback
// ---------------------------------------------------------------------------
#if __has_builtin(__builtin_amdgcn_global_load_async_to_lds_b128)
#define HAVE_ASYNC_LDS 1
__device__ __forceinline__ void async_copy_b128(const u16* g, u16* lds) {
  __builtin_amdgcn_global_load_async_to_lds_b128(
      (__attribute__((address_space(1))) v4i*)(g),
      (__attribute__((address_space(3))) v4i*)(lds), 0, 0);
}
#else
#define HAVE_ASYNC_LDS 0
__device__ __forceinline__ void async_copy_b128(const u16* g, u16* lds) {
  *(uint4*)lds = *(const uint4*)g;   // synchronous fallback
}
#endif

#if __has_builtin(__builtin_amdgcn_s_wait_asynccnt)
#define WAIT_ASYNC(n) __builtin_amdgcn_s_wait_asynccnt(n)
#else
#define WAIT_ASYNC(n) asm volatile("s_wait_asynccnt %0" :: "i"(n) : "memory")
#endif

// Problem constants
#define BATCH 64
#define TSEQ  128
#define EMB   256
#define HID   512
#define G3H   1536            // 3*HID
#define ROWS  (BATCH*TSEQ)    // 8192
#define K0P   288             // E+1=257 padded to mult of 32
#define KB0   (K0P/32)        // 9
#define KBH   (HID/32)        // 16
#define VOC   10000
#define NFCP  10112           // 10001 padded to mult of 128

// ---------------------------------------------------------------------------
// Pack f32 [N,K] -> bf16 fragment-major [Ntiles*Kb*512] (zero padded)
// Destination-major iteration -> coalesced stores.
// ---------------------------------------------------------------------------
__global__ void pack_bf16_sw_kernel(const float* __restrict__ src, u16* __restrict__ dst,
                                    int N, int K, int Ntiles, int Kb) {
  int total = Ntiles * Kb * 512;
  for (int i = blockIdx.x * blockDim.x + threadIdx.x; i < total;
       i += gridDim.x * blockDim.x) {
    int tile = i >> 9, w = i & 511;
    int lane = w >> 4, e = w & 15;
    int rt = tile / Kb, kb = tile % Kb;
    int v = e >> 1, j = e & 1;
    int kk = ((v & 4) << 2) + ((lane >> 4) << 3) + ((v & 3) << 1) + j;
    int n = rt * 16 + (lane & 15);
    int k = kb * 32 + kk;
    float val = (n < N && k < K) ? src[(size_t)n * K + k] : 0.0f;
    dst[i] = f2bf(val);
  }
}

// ---------------------------------------------------------------------------
// Build x0 = [emb[loc] | time | 0-pad] bf16, fragment-major [512 tiles x 9 kb]
// ---------------------------------------------------------------------------
__global__ void build_x0_sw_kernel(const int* __restrict__ loc,
                                   const float* __restrict__ times,
                                   const float* __restrict__ emb,
                                   u16* __restrict__ x0) {
  int total = (ROWS / 16) * KB0 * 512;
  for (int i = blockIdx.x * blockDim.x + threadIdx.x; i < total;
       i += gridDim.x * blockDim.x) {
    int tile = i >> 9, w = i & 511;
    int lane = w >> 4, e = w & 15;
    int rt = tile / KB0, kb = tile % KB0;
    int v = e >> 1, j = e & 1;
    int kk = ((v & 4) << 2) + ((lane >> 4) << 3) + ((v & 3) << 1) + j;
    int r = rt * 16 + (lane & 15);
    int c = kb * 32 + kk;
    float val;
    if (c < EMB)       val = emb[(size_t)loc[r] * EMB + c];
    else if (c == EMB) val = times[r];
    else               val = 0.0f;
    x0[i] = f2bf(val);
  }
}

// ---------------------------------------------------------------------------
// WMMA GEMM on fragment-major operands:
//   C[M,N] (f32) = A[M,K] @ W[N,K]^T + bias
// Block tile 64x128, 256 threads = 8 waves, wave -> 16x64 (4 wmma tiles).
// Async double-buffered LDS staging: 3 async b128 per wave per k-tile,
// pipelined with s_wait_asynccnt so DMA of tile k+1 overlaps WMMA of tile k.
// mode 0: plain store; mode 1: FC epilogue (logits | sigmoid time)
// ---------------------------------------------------------------------------
__global__ __launch_bounds__(256)
void gemm_bias_kernel(const u16* __restrict__ Asw, int kb_total,
                      const u16* __restrict__ Wsw,
                      const float* __restrict__ bias,
                      float* __restrict__ C, int ldc,
                      int mode, float* __restrict__ time_out) {
  __shared__ alignas(16) u16 AsF[2][4 * 512];   //  8 KB
  __shared__ alignas(16) u16 BsF[2][8 * 512];   // 16 KB

  const int tid  = threadIdx.x;
  const int lane = tid & 31;
  const int wid  = tid >> 5;
  const int mt0  = blockIdx.y * 4;   // first m-tile of block
  const int nt0  = blockIdx.x * 8;   // first n-tile of block
  const int mt   = wid & 3;          // wave m-tile
  const int ng   = wid >> 2;         // wave n-group (4 tiles each)

  v8f acc[4];
#pragma unroll
  for (int j = 0; j < 4; ++j)
#pragma unroll
    for (int e = 0; e < 8; ++e) acc[j][e] = 0.0f;

  auto stage = [&](int kb, int buf) {
    // A: 4 tiles * 64 chunks of b128
    {
      int at = tid >> 6, ac = tid & 63;
      const u16* g = Asw + ((size_t)(mt0 + at) * kb_total + kb) * 512 + ac * 8;
      async_copy_b128(g, &AsF[buf][at * 512 + ac * 8]);
    }
    // B: 8 tiles * 64 chunks
#pragma unroll
    for (int rep = 0; rep < 2; ++rep) {
      int idx = tid + rep * 256;
      int bt = idx >> 6, bc = idx & 63;
      const u16* g = Wsw + ((size_t)(nt0 + bt) * kb_total + kb) * 512 + bc * 8;
      async_copy_b128(g, &BsF[buf][bt * 512 + bc * 8]);
    }
  };

  stage(0, 0);
  for (int kb = 0; kb < kb_total; ++kb) {
    int buf = kb & 1;
    bool more = (kb + 1 < kb_total);
    if (more) stage(kb + 1, buf ^ 1);        // overlap DMA with compute
    if (more) { WAIT_ASYNC(3); } else { WAIT_ASYNC(0); }
    __syncthreads();                          // tile kb visible to all waves

    FragAB a;
    {
      const uint4* p = (const uint4*)&AsF[buf][mt * 512 + lane * 16];
      a.q[0] = p[0]; a.q[1] = p[1];
    }
#pragma unroll
    for (int j = 0; j < 4; ++j) {
      FragAB b;
      const uint4* p = (const uint4*)&BsF[buf][(ng * 4 + j) * 512 + lane * 16];
      b.q[0] = p[0]; b.q[1] = p[1];
      acc[j] = __builtin_amdgcn_wmma_f32_16x16x32_bf16(
          false, a.v, false, b.v, (short)0, acc[j], false, false);
    }
    __syncthreads();                          // buf may be overwritten next iter
  }

#pragma unroll
  for (int j = 0; j < 4; ++j) {
    int nn = (nt0 + ng * 4 + j) * 16 + (lane & 15);
#pragma unroll
    for (int v = 0; v < 8; ++v) {
      int mm = (mt0 + mt) * 16 + v + ((lane >> 4) << 3);
      float val = acc[j][v];
      if (mode == 0) {
        C[(size_t)mm * ldc + nn] = val + bias[nn];
      } else {
        if (nn < VOC)       C[(size_t)mm * VOC + nn] = val + bias[nn];
        else if (nn == VOC) time_out[mm] = sigmoidf_(val + bias[nn]);
      }
    }
  }
}

// ---------------------------------------------------------------------------
// One GRU timestep. Grid = 8 blocks; block owns 64 hidden cols.
// h_prev [64,512] and w_hh [1536,512] are fragment-major bf16 -> coalesced
// b128 fragment loads straight from L2; gh exchanged through LDS.
// ---------------------------------------------------------------------------
__global__ __launch_bounds__(256)
void gru_step_kernel(const float* __restrict__ xproj,  // [8192,1536], b_ih fused
                     const float* __restrict__ b_hh,   // [1536]
                     const u16*  __restrict__ h_prev,  // frag-major [4x16x512]
                     const u16*  __restrict__ w_hh,    // frag-major [96x16x512]
                     u16* __restrict__ h_next,         // frag-major [4x16x512]
                     u16* __restrict__ h_seq,          // frag-major [512x16x512]
                     int t, int relu_seq) {
  __shared__ float gh[3][BATCH][66];

  const int tid  = threadIdx.x;
  const int lane = tid & 31;
  const int wid  = tid >> 5;
  const int n0   = blockIdx.x * 64;
  const int mt   = wid & 3;
  const int half = wid >> 2;   // nt = half*6 .. half*6+5

  v8f acc[6];
#pragma unroll
  for (int j = 0; j < 6; ++j)
#pragma unroll
    for (int e = 0; e < 8; ++e) acc[j][e] = 0.0f;

  for (int kb = 0; kb < KBH; ++kb) {
    FragAB a;
    {
      const uint4* p =
          (const uint4*)&h_prev[((size_t)mt * KBH + kb) * 512 + lane * 16];
      a.q[0] = p[0]; a.q[1] = p[1];
    }
#pragma unroll
    for (int j = 0; j < 6; ++j) {
      int nt = half * 6 + j;
      int g = nt >> 2, ct = nt & 3;
      int rt = g * 32 + (n0 >> 4) + ct;      // W row-tile
      FragAB b;
      const uint4* p =
          (const uint4*)&w_hh[((size_t)rt * KBH + kb) * 512 + lane * 16];
      b.q[0] = p[0]; b.q[1] = p[1];
      acc[j] = __builtin_amdgcn_wmma_f32_16x16x32_bf16(
          false, a.v, false, b.v, (short)0, acc[j], false, false);
    }
  }

#pragma unroll
  for (int j = 0; j < 6; ++j) {
    int nt = half * 6 + j;
    int g = nt >> 2, ct = nt & 3;
    int col = ct * 16 + (lane & 15);
#pragma unroll
    for (int v = 0; v < 8; ++v) {
      int m = mt * 16 + v + ((lane >> 4) << 3);
      gh[g][m][col] = acc[j][v];
    }
  }
  __syncthreads();

  // fused gate math: 64x64 elems, 16 per thread
#pragma unroll 4
  for (int i = 0; i < 16; ++i) {
    int idx = i * 256 + tid;
    int b = idx >> 6, jj = idx & 63;
    int hcol = n0 + jj;
    size_t row = (size_t)b * TSEQ + t;
    const float* xp = xproj + row * G3H;
    float xr = xp[hcol];
    float xz = xp[HID + hcol];
    float xn = xp[2 * HID + hcol];
    float hr = gh[0][b][jj] + b_hh[hcol];
    float hz = gh[1][b][jj] + b_hh[HID + hcol];
    float hn = gh[2][b][jj] + b_hh[2 * HID + hcol];
    float r = sigmoidf_(xr + hr);
    float z = sigmoidf_(xz + hz);
    float n = tanhf(xn + r * hn);
    size_t hoff = sw_off(b, hcol, KBH);
    float hp = bf2f(h_prev[hoff]);
    float h = (1.0f - z) * n + z * hp;
    h_next[hoff] = f2bf(h);
    float hs = relu_seq ? fmaxf(h, 0.0f) : h;
    h_seq[sw_off((int)row, hcol, KBH)] = f2bf(hs);
  }
}

// ---------------------------------------------------------------------------
// In-place log-softmax over rows of [8192, 10000]
// ---------------------------------------------------------------------------
__global__ __launch_bounds__(256)
void log_softmax_kernel(float* __restrict__ out) {
  __shared__ float redmax[8];
  __shared__ float redsum[8];
  const int tid = threadIdx.x, lane = tid & 31, wid = tid >> 5;
  float* p = out + (size_t)blockIdx.x * VOC;

  float mx = -INFINITY;
  for (int i = tid; i < VOC; i += 256) mx = fmaxf(mx, p[i]);
#pragma unroll
  for (int o = 16; o > 0; o >>= 1) mx = fmaxf(mx, __shfl_xor(mx, o, 32));
  if (lane == 0) redmax[wid] = mx;
  __syncthreads();
  mx = fmaxf(fmaxf(fmaxf(redmax[0], redmax[1]), fmaxf(redmax[2], redmax[3])),
             fmaxf(fmaxf(redmax[4], redmax[5]), fmaxf(redmax[6], redmax[7])));

  float s = 0.0f;
  for (int i = tid; i < VOC; i += 256) s += __expf(p[i] - mx);
#pragma unroll
  for (int o = 16; o > 0; o >>= 1) s += __shfl_xor(s, o, 32);
  if (lane == 0) redsum[wid] = s;
  __syncthreads();
  s = (redsum[0] + redsum[1]) + (redsum[2] + redsum[3]) +
      (redsum[4] + redsum[5]) + (redsum[6] + redsum[7]);
  float lse = mx + __logf(s);
  for (int i = tid; i < VOC; i += 256) p[i] = p[i] - lse;
}

// ---------------------------------------------------------------------------
// Host launcher
// ---------------------------------------------------------------------------
#define GBN 128
#define GBM 64

static inline size_t align256(size_t x) { return (x + 255) & ~(size_t)255; }

extern "C" void kernel_launch(void* const* d_in, const int* in_sizes, int n_in,
                              void* d_out, int out_size, void* d_ws, size_t ws_size,
                              hipStream_t stream) {
  const int*   loc   = (const int*)  d_in[0];
  const float* times = (const float*)d_in[1];
  const float* emb   = (const float*)d_in[2];
  const float* w_ih0 = (const float*)d_in[3];
  const float* w_hh0 = (const float*)d_in[4];
  const float* b_ih0 = (const float*)d_in[5];
  const float* b_hh0 = (const float*)d_in[6];
  const float* w_ih1 = (const float*)d_in[7];
  const float* w_hh1 = (const float*)d_in[8];
  const float* b_ih1 = (const float*)d_in[9];
  const float* b_hh1 = (const float*)d_in[10];
  const float* fc_w  = (const float*)d_in[11];
  const float* fc_b  = (const float*)d_in[12];

  float* loc_out  = (float*)d_out;                         // [8192,10000]
  float* time_out = (float*)d_out + (size_t)ROWS * VOC;    // [8192]

  char* ws = (char*)d_ws;
  auto take = [&](size_t bytes) { char* p = ws; ws += align256(bytes); return p; };
  u16*   w_ih0p = (u16*)  take((size_t)G3H  * K0P * 2);
  u16*   w_hh0p = (u16*)  take((size_t)G3H  * HID * 2);
  u16*   w_ih1p = (u16*)  take((size_t)G3H  * HID * 2);
  u16*   w_hh1p = (u16*)  take((size_t)G3H  * HID * 2);
  u16*   fc_wp  = (u16*)  take((size_t)NFCP * HID * 2);
  u16*   x0     = (u16*)  take((size_t)ROWS * K0P * 2);
  float* xproj  = (float*)take((size_t)ROWS * G3H * 4);    // reused by both layers
  u16*   h1seq  = (u16*)  take((size_t)ROWS * HID * 2);
  u16*   h2seq  = (u16*)  take((size_t)ROWS * HID * 2);    // relu'd, feeds FC
  u16*   hA     = (u16*)  take((size_t)BATCH * HID * 2);
  u16*   hB     = (u16*)  take((size_t)BATCH * HID * 2);

  // 1) pack weights to fragment-major bf16
  pack_bf16_sw_kernel<<<1024, 256, 0, stream>>>(w_ih0, w_ih0p, G3H, EMB + 1,
                                                G3H / 16, KB0);
  pack_bf16_sw_kernel<<<1024, 256, 0, stream>>>(w_hh0, w_hh0p, G3H, HID,
                                                G3H / 16, KBH);
  pack_bf16_sw_kernel<<<1024, 256, 0, stream>>>(w_ih1, w_ih1p, G3H, HID,
                                                G3H / 16, KBH);
  pack_bf16_sw_kernel<<<1024, 256, 0, stream>>>(w_hh1, w_hh1p, G3H, HID,
                                                G3H / 16, KBH);
  pack_bf16_sw_kernel<<<2048, 256, 0, stream>>>(fc_w,  fc_wp,  VOC + 1, HID,
                                                NFCP / 16, KBH);

  // 2) embedding gather + time concat (fragment-major)
  build_x0_sw_kernel<<<1024, 256, 0, stream>>>(loc, times, emb, x0);

  // 3) layer-0 input projection
  {
    dim3 g(G3H / GBN, ROWS / GBM);
    gemm_bias_kernel<<<g, 256, 0, stream>>>(x0, KB0, w_ih0p, b_ih0,
                                            xproj, G3H, 0, nullptr);
  }

  // 4) layer-0 recurrence
  (void)hipMemsetAsync(hA, 0, (size_t)BATCH * HID * 2, stream);
  for (int t = 0; t < TSEQ; ++t) {
    const u16* hp = (t & 1) ? hB : hA;
    u16*       hn = (t & 1) ? hA : hB;
    gru_step_kernel<<<HID / 64, 256, 0, stream>>>(xproj, b_hh0, hp, w_hh0p,
                                                  hn, h1seq, t, 0);
  }

  // 5) layer-1 input projection
  {
    dim3 g(G3H / GBN, ROWS / GBM);
    gemm_bias_kernel<<<g, 256, 0, stream>>>(h1seq, KBH, w_ih1p, b_ih1,
                                            xproj, G3H, 0, nullptr);
  }

  // 6) layer-1 recurrence (h_seq gets relu for the FC)
  (void)hipMemsetAsync(hA, 0, (size_t)BATCH * HID * 2, stream);
  for (int t = 0; t < TSEQ; ++t) {
    const u16* hp = (t & 1) ? hB : hA;
    u16*       hn = (t & 1) ? hA : hB;
    gru_step_kernel<<<HID / 64, 256, 0, stream>>>(xproj, b_hh1, hp, w_hh1p,
                                                  hn, h2seq, t, 1);
  }

  // 7) FC: relu(h2) @ fc_w^T + fc_b, split epilogue
  {
    dim3 g(NFCP / GBN, ROWS / GBM);
    gemm_bias_kernel<<<g, 256, 0, stream>>>(h2seq, KBH, fc_wp, fc_b,
                                            loc_out, 0, 1, time_out);
  }

  // 8) in-place log-softmax
  log_softmax_kernel<<<ROWS, 256, 0, stream>>>(loc_out);
}
